// DGCNN_40716289966807
// MI455X (gfx1250) — matrix operand order repaired and checked by reading, not compiled, not run on hardware
//
#include <hip/hip_runtime.h>

typedef __attribute__((ext_vector_type(16))) _Float16 v16h;
typedef __attribute__((ext_vector_type(8)))  _Float16 v8h;
typedef __attribute__((ext_vector_type(8)))  float    v8f;

constexpr int   kK   = 20;
constexpr int   kN   = 4096;
constexpr int   kB   = 4;
constexpr int   kGW  = 8;      // waves per block in WMMA kernels
constexpr float kEps = 1e-5f;

__device__ __forceinline__ float lrelu(float v) { return v >= 0.f ? v : 0.2f * v; }

#if defined(__gfx1250__) && __has_builtin(__builtin_amdgcn_global_load_async_to_lds_b128) && \
    __has_builtin(__builtin_amdgcn_s_wait_asynccnt)
#define USE_ASYNC_LDS 1
// builtin prototype (from hipcc diagnostic): src is int4* in AS(1) (__device__),
// dst is int4* in AS(3) (__shared__), then imm offset, imm cpol.
typedef int v4i_vs __attribute__((__vector_size__(16)));
typedef __attribute__((address_space(1))) v4i_vs glb_v4i;
typedef __attribute__((address_space(3))) v4i_vs lds_v4i;
#else
#define USE_ASYNC_LDS 0
#endif

// Stage a contiguous global panel (multiple of 8 halfs, 16B aligned) into LDS,
// using CDNA5 async global->LDS copies when available (ASYNCcnt path).
__device__ __forceinline__ void stage_panel(const _Float16* __restrict__ gsrc,
                                            _Float16* sdst, int halfs,
                                            int tid, int nthreads) {
#if USE_ASYNC_LDS
  for (int i = tid * 8; i < halfs; i += nthreads * 8) {
    __builtin_amdgcn_global_load_async_to_lds_b128(
        (glb_v4i*)(gsrc + i), (lds_v4i*)(sdst + i), 0, 0);
  }
  __builtin_amdgcn_s_wait_asynccnt(0);
#else
  for (int i = tid * 8; i < halfs; i += nthreads * 8)
    *(v8h*)(sdst + i) = *(const v8h*)(gsrc + i);
#endif
}

// ---- WMMA fragment loaders (wave32, 16x16x32 f16, per ISA 7.12.2) ---------
// A (16x32, MxK): lane m=lane&15; halfs 0..7  = A[m, k0+8*half + 0..7]
//                                halfs 8..15 = A[m, k0+16+8*half + 0..7]
__device__ __forceinline__ v16h load_a_frag(const _Float16* __restrict__ row,
                                            int k0, int half) {
  const v8h p0 = *(const v8h*)(row + k0 + 8 * half);
  const v8h p1 = *(const v8h*)(row + k0 + 8 * half + 16);
  v16h a;
#pragma unroll
  for (int j = 0; j < 8; ++j) { a[j] = p0[j]; a[8 + j] = p1[j]; }
  return a;
}
// B (32x16, KxN): lane n=lane&15; halfs 0..7 = B[k0 + 8*half + 0..7, n],
// halfs 8..15 = B[k0 + 16 + 8*half + 0..7, n] (row n of W, same order as A).
__device__ __forceinline__ v16h load_b_frag(const _Float16* row, int k0, int half) {
  const v8h p0 = *(const v8h*)(row + k0 + 8 * half);
  const v8h p1 = *(const v8h*)(row + k0 + 8 * half + 16);
  v16h b;
#pragma unroll
  for (int j = 0; j < 8; ++j) { b[j] = p0[j]; b[8 + j] = p1[j]; }
  return b;
}

// ---- prep kernels ---------------------------------------------------------
__global__ void wpad_f16_kernel(const float* __restrict__ w, _Float16* __restrict__ wp,
                                int O, int Kin, int Kpad) {
  int t = blockIdx.x * blockDim.x + threadIdx.x;
  if (t >= O * Kpad) return;
  int k = t % Kpad, o = t / Kpad;
  wp[t] = (_Float16)((k < Kin) ? w[(size_t)o * Kin + k] : 0.f);
}

// split fp32 feature into hi/lo f16 pair (error-compensated WMMA input)
__global__ void split_hilo_kernel(const float* __restrict__ feat, // [N,C] batch base
                                  _Float16* __restrict__ hi, _Float16* __restrict__ lo,
                                  int N, int C, int Kpad) {
  int t = blockIdx.x * blockDim.x + threadIdx.x;
  if (t >= N * Kpad) return;
  int k = t % Kpad, n = t / Kpad;
  float v = (k < C) ? feat[(size_t)n * C + k] : 0.f;
  _Float16 h = (_Float16)v;
  hi[t] = h;
  lo[t] = (_Float16)(v - (float)h);
}

__global__ void sqnorm_kernel(const float* __restrict__ feat, float* __restrict__ sq,
                              int N, int C) {
  int n = blockIdx.x * blockDim.x + threadIdx.x;
  if (n >= N) return;
  const float* f = feat + (size_t)n * C;
  float s = 0.f;
  for (int c = 0; c < C; ++c) s += f[c] * f[c];
  sq[n] = s;
}

// ---- Gram / pairwise distance via error-compensated f16 WMMA --------------
// grid = (N/16, N/(16*kGW)), block = 256 (8 waves). All waves in a block share
// one candidate (column) tile staged in LDS; each wave owns one query tile.
__global__ __launch_bounds__(256) void gram_dist_kernel(
    const _Float16* __restrict__ hi, const _Float16* __restrict__ lo,
    const float* __restrict__ sq, float* __restrict__ dist, int N, int Kpad) {
  __shared__ alignas(32) _Float16 sBH[16 * 128];
  __shared__ alignas(32) _Float16 sBL[16 * 128];
  const int tid  = threadIdx.x;
  const int lane = tid & 31;
  const int wave = tid >> 5;
  const int half = lane >> 4;
  const int l    = lane & 15;
  const int colTile = blockIdx.x;
  const int rowTile = blockIdx.y * kGW + wave;

  stage_panel(hi + (size_t)colTile * 16 * Kpad, sBH, 16 * Kpad, tid, 256);
  stage_panel(lo + (size_t)colTile * 16 * Kpad, sBL, 16 * Kpad, tid, 256);
  __syncthreads();

  const int row = rowTile * 16 + l;
  const int col = colTile * 16 + l;
  const _Float16* aH = hi + (size_t)row * Kpad;
  const _Float16* aL = lo + (size_t)row * Kpad;
  const _Float16* bH = sBH + (size_t)l * Kpad;   // LDS reads -> ds_load_b128
  const _Float16* bL = sBL + (size_t)l * Kpad;
  v8f acc = {0.f, 0.f, 0.f, 0.f, 0.f, 0.f, 0.f, 0.f};
  for (int k0 = 0; k0 < Kpad; k0 += 32) {
    v16h ah = load_a_frag(aH, k0, half);
    v16h al = load_a_frag(aL, k0, half);
    v16h bh = load_b_frag(bH, k0, half);
    v16h bl = load_b_frag(bL, k0, half);
    acc = __builtin_amdgcn_wmma_f32_16x16x32_f16(false, ah, false, bh, (short)0, acc, false, false);
    acc = __builtin_amdgcn_wmma_f32_16x16x32_f16(false, ah, false, bl, (short)0, acc, false, false);
    acc = __builtin_amdgcn_wmma_f32_16x16x32_f16(false, al, false, bh, (short)0, acc, false, false);
  }
  const float sqj = sq[col];
#pragma unroll
  for (int r = 0; r < 8; ++r) {
    int m = rowTile * 16 + r + 8 * half;   // C/D layout: vgpr r -> row r (+8 for hi lanes)
    dist[(size_t)m * N + col] = sq[m] + sqj - 2.f * acc[r];
  }
}

// ---- streaming top-k (k=20, smallest distance, stable on ties) ------------
__global__ void topk_kernel(const float* __restrict__ dist, int* __restrict__ idx, int N) {
  int n = blockIdx.x * blockDim.x + threadIdx.x;
  if (n >= N) return;
  float bd[kK]; int bi[kK];
#pragma unroll
  for (int t = 0; t < kK; ++t) { bd[t] = 3.0e38f; bi[t] = 0; }
  const float* dr = dist + (size_t)n * N;
  for (int j = 0; j < N; ++j) {
    if ((j & 63) == 0) __builtin_prefetch(dr + j + 256, 0, 0);
    float d = dr[j];
    if (d < bd[kK - 1]) {
      float dd = d; int ii = j;
#pragma unroll
      for (int t = 0; t < kK; ++t) {
        if (dd < bd[t]) {
          float tf = bd[t]; bd[t] = dd; dd = tf;
          int   ti = bi[t]; bi[t] = ii; ii = ti;
        }
      }
    }
  }
  int* op = idx + (size_t)n * kK;
#pragma unroll
  for (int t = 0; t < kK; ++t) op[t] = bi[t];
}

// ---- edge-feature gathers (write f16, zero-padded to WMMA K granularity) --
__global__ void gather_l1_kernel(const float* __restrict__ xyz, const float* __restrict__ pf,
                                 const int* __restrict__ idx, _Float16* __restrict__ h, int N) {
  int t = blockIdx.x * blockDim.x + threadIdx.x;
  if (t >= N * kK) return;
  int n = t / kK, kk = t % kK;
  int j = idx[(size_t)n * kK + kk];
  _Float16* hr = h + (size_t)t * 32;
#pragma unroll
  for (int c = 0; c < 3; ++c) {
    hr[c]     = (_Float16)xyz[(size_t)j * 3 + c];
    hr[3 + c] = (_Float16)pf [(size_t)j * 3 + c];
    hr[6 + c] = (_Float16)xyz[(size_t)n * 3 + c];
    hr[9 + c] = (_Float16)pf [(size_t)n * 3 + c];
  }
#pragma unroll
  for (int c = 12; c < 32; ++c) hr[c] = (_Float16)0.f;
}

__global__ void gather_edge_kernel(const float* __restrict__ feat, const int* __restrict__ idx,
                                   _Float16* __restrict__ h, int N, int C) {
  int t = blockIdx.x * blockDim.x + threadIdx.x;
  if (t >= N * kK * C) return;
  int c = t % C, r = t / C;
  int n = r / kK, kk = r % kK;
  int j = idx[(size_t)n * kK + kk];
  _Float16* hr = h + (size_t)r * (2 * C);
  hr[c]     = (_Float16)feat[(size_t)j * C + c];
  hr[C + c] = (_Float16)feat[(size_t)n * C + c];
}

// ---- WMMA GEMM + BN + LeakyReLU epilogue ----------------------------------
// Y[M,O] = A[M,Kp] * Wp[O,Kp]^T ; grid = (O/16, M/(16*kGW)), block = 256.
// The 16-row weight panel is staged to LDS once per block and shared by 8 waves.
__global__ __launch_bounds__(256) void gemm_bn_lrelu_kernel(
    const _Float16* __restrict__ A, const _Float16* __restrict__ Wp,
    const float* __restrict__ g, const float* __restrict__ bt,
    const float* __restrict__ rm, const float* __restrict__ rv,
    float* __restrict__ Y, int M, int Kp, int O) {
  __shared__ alignas(32) _Float16 sW[16 * 512];
  const int tid  = threadIdx.x;
  const int lane = tid & 31;
  const int wave = tid >> 5;
  const int half = lane >> 4;
  const int l    = lane & 15;
  const int tileO = blockIdx.x;
  const int tileM = blockIdx.y * kGW + wave;

  stage_panel(Wp + (size_t)tileO * 16 * Kp, sW, 16 * Kp, tid, 256);
  __syncthreads();

  const int mrow = tileM * 16 + l;
  const int ocol = tileO * 16 + l;
  const _Float16* arow = A  + (size_t)mrow * Kp;
  const _Float16* wrow = sW + (size_t)l * Kp;   // LDS reads -> ds_load_b128
  v8f acc = {0.f, 0.f, 0.f, 0.f, 0.f, 0.f, 0.f, 0.f};
  for (int k0 = 0; k0 < Kp; k0 += 32) {
    v16h a = load_a_frag(arow, k0, half);
    v16h w = load_b_frag(wrow, k0, half);
    acc = __builtin_amdgcn_wmma_f32_16x16x32_f16(false, a, false, w, (short)0, acc, false, false);
  }
  float sc = g[ocol] * rsqrtf(rv[ocol] + kEps);
  float sh = bt[ocol] - rm[ocol] * sc;
#pragma unroll
  for (int r = 0; r < 8; ++r) {
    int m = tileM * 16 + r + 8 * half;
    Y[(size_t)m * O + ocol] = lrelu(acc[r] * sc + sh);
  }
}

// ---- max over k neighbors -------------------------------------------------
__global__ void max_over_k_kernel(const float* __restrict__ Y, float* __restrict__ X,
                                  int N, int O) {
  int t = blockIdx.x * blockDim.x + threadIdx.x;
  if (t >= N * O) return;
  int o = t % O, n = t / O;
  const float* yb = Y + ((size_t)n * kK) * O + o;
  float m = yb[0];
  for (int kk = 1; kk < kK; ++kk) m = fmaxf(m, yb[(size_t)kk * O]);
  X[(size_t)n * O + o] = m;
}

// ---- concat x1..x4 -> f16 [B*N,512] ---------------------------------------
__global__ void pack_xcat_kernel(const float* __restrict__ x1, const float* __restrict__ x2,
                                 const float* __restrict__ x3, const float* __restrict__ x4,
                                 _Float16* __restrict__ xc, int BN) {
  int t = blockIdx.x * blockDim.x + threadIdx.x;
  if (t >= BN * 512) return;
  int c = t % 512, r = t / 512;
  float v;
  if (c < 64)       v = x1[(size_t)r * 64  + c];
  else if (c < 128) v = x2[(size_t)r * 64  + (c - 64)];
  else if (c < 256) v = x3[(size_t)r * 128 + (c - 128)];
  else              v = x4[(size_t)r * 256 + (c - 256)];
  xc[t] = (_Float16)v;
}

// ---- global max + mean pool over N ----------------------------------------
__global__ void global_pool_kernel(const float* __restrict__ y5, float* __restrict__ z, int N) {
  int t = blockIdx.x * blockDim.x + threadIdx.x;
  if (t >= kB * 512) return;
  int o = t % 512, b = t / 512;
  const float* base = y5 + ((size_t)b * N) * 512 + o;
  float mx = -3.0e38f, sum = 0.f;
  for (int n = 0; n < N; ++n) {
    float v = base[(size_t)n * 512];
    mx = fmaxf(mx, v);
    sum += v;
  }
  z[(size_t)b * 1024 + o]       = mx;
  z[(size_t)b * 1024 + 512 + o] = sum * (1.f / (float)N);
}

// ---- tiny FC head (B=4 rows; scalar is fine) ------------------------------
__global__ void fc_kernel(const float* __restrict__ zin, const float* __restrict__ w,
                          const float* __restrict__ bias,
                          const float* __restrict__ g, const float* __restrict__ bt,
                          const float* __restrict__ rm, const float* __restrict__ rv,
                          float* __restrict__ out, int Bn, int In, int On, int mode) {
  int t = blockIdx.x * blockDim.x + threadIdx.x;
  if (t >= Bn * On) return;
  int o = t % On, r = t / On;
  const float* zr = zin + (size_t)r * In;
  const float* wr = w   + (size_t)o * In;
  float s = 0.f;
  for (int i = 0; i < In; ++i) s += zr[i] * wr[i];
  if (bias) s += bias[o];
  if (mode) {
    float sc = g[o] * rsqrtf(rv[o] + kEps);
    s = lrelu(s * sc + (bt[o] - rm[o] * sc));
  }
  out[t] = s;
}

// ---------------------------------------------------------------------------
extern "C" void kernel_launch(void* const* d_in, const int* in_sizes, int n_in,
                              void* d_out, int out_size, void* d_ws, size_t ws_size,
                              hipStream_t stream) {
  (void)in_sizes; (void)n_in; (void)out_size; (void)ws_size;
  const float* xyz = (const float*)d_in[0];
  const float* pf  = (const float*)d_in[1];
  const float* w1  = (const float*)d_in[2];
  const float* w2  = (const float*)d_in[3];
  const float* w3  = (const float*)d_in[4];
  const float* w4  = (const float*)d_in[5];
  const float* w5  = (const float*)d_in[6];
  const float* bn[7][4];
  for (int i = 0; i < 7; ++i)
    for (int j = 0; j < 4; ++j) bn[i][j] = (const float*)d_in[7 + i * 4 + j];
  const float* l1w = (const float*)d_in[35];
  const float* l2w = (const float*)d_in[36];
  const float* l2b = (const float*)d_in[37];
  const float* l3w = (const float*)d_in[38];
  const float* l3b = (const float*)d_in[39];
  float* out = (float*)d_out;

  char* ws = (char*)d_ws;
  size_t off = 0;
  auto take = [&](size_t bytes) -> char* {
    char* p = ws + off;
    off += (bytes + 255) & ~(size_t)255;
    return p;
  };
  _Float16* wp1  = (_Float16*)take((size_t)64  * 32  * 2);
  _Float16* wp2  = (_Float16*)take((size_t)64  * 128 * 2);
  _Float16* wp3  = (_Float16*)take((size_t)128 * 128 * 2);
  _Float16* wp4  = (_Float16*)take((size_t)256 * 256 * 2);
  _Float16* wp5  = (_Float16*)take((size_t)512 * 512 * 2);
  float*    sq   = (float*)take((size_t)kN * 4);
  _Float16* hib  = (_Float16*)take((size_t)kN * 128 * 2);
  _Float16* lob  = (_Float16*)take((size_t)kN * 128 * 2);
  int*      idxb = (int*)take((size_t)kB * kN * kK * 4);
  float*    x1   = (float*)take((size_t)kB * kN * 64  * 4);
  float*    x2   = (float*)take((size_t)kB * kN * 64  * 4);
  float*    x3   = (float*)take((size_t)kB * kN * 128 * 4);
  float*    x4   = (float*)take((size_t)kB * kN * 256 * 4);
  _Float16* xcat = (_Float16*)take((size_t)kB * kN * 512 * 2);
  float*    y5   = (float*)take((size_t)kB * kN * 512 * 4);
  float*    zb   = (float*)take((size_t)kB * 1024 * 4);
  float*    z1   = (float*)take((size_t)kB * 512 * 4);
  float*    z2   = (float*)take((size_t)kB * 256 * 4);
  // dist (N*N f32 = 64MB) and per-layer y (N*k*O f32 <= 80MB) have disjoint
  // lifetimes within a (batch,layer) step -> overlay them in one region.
  float*    distY = (float*)take((size_t)kN * kK * 256 * 4);
  _Float16* hbuf  = (_Float16*)take((size_t)kN * kK * 512 * 2);

  auto g1d = [](int total, int bs) { return dim3((unsigned)((total + bs - 1) / bs)); };

  // weight conversion / padding (tiny)
  wpad_f16_kernel<<<g1d(64 * 32, 256),   256, 0, stream>>>(w1, wp1, 64, 12, 32);
  wpad_f16_kernel<<<g1d(64 * 128, 256),  256, 0, stream>>>(w2, wp2, 64, 128, 128);
  wpad_f16_kernel<<<g1d(128 * 128, 256), 256, 0, stream>>>(w3, wp3, 128, 128, 128);
  wpad_f16_kernel<<<g1d(256 * 256, 256), 256, 0, stream>>>(w4, wp4, 256, 256, 256);
  wpad_f16_kernel<<<g1d(512 * 512, 256), 256, 0, stream>>>(w5, wp5, 512, 512, 512);

  const dim3 ggram(kN / 16, kN / (16 * kGW));   // (256, 32)
  const int  Mrows = kN * kK;                   // 81920, multiple of 16*kGW

  auto run_edge = [&](int b, const float* feat, int C, int Kfeat,
                      const _Float16* wp, int Kh, int O, int li, float* xout) {
    const float* fb = feat + (size_t)b * kN * C;
    split_hilo_kernel<<<g1d(kN * Kfeat, 256), 256, 0, stream>>>(fb, hib, lob, kN, C, Kfeat);
    sqnorm_kernel<<<g1d(kN, 256), 256, 0, stream>>>(fb, sq, kN, C);
    gram_dist_kernel<<<ggram, 256, 0, stream>>>(hib, lob, sq, distY, kN, Kfeat);
    int* ib = idxb + (size_t)b * kN * kK;
    topk_kernel<<<g1d(kN, 256), 256, 0, stream>>>(distY, ib, kN);
    gather_edge_kernel<<<g1d(kN * kK * C, 256), 256, 0, stream>>>(fb, ib, hbuf, kN, C);
    dim3 gm(O / 16, Mrows / (16 * kGW));
    gemm_bn_lrelu_kernel<<<gm, 256, 0, stream>>>(hbuf, wp, bn[li][0], bn[li][1], bn[li][2],
                                                 bn[li][3], distY, Mrows, Kh, O);
    max_over_k_kernel<<<g1d(kN * O, 256), 256, 0, stream>>>(distY, xout + (size_t)b * kN * O, kN, O);
  };

  for (int b = 0; b < kB; ++b) {
    // ---- EdgeConv 1: graph from xyz, features = [xyz, point_fea] ----------
    const float* xb = xyz + (size_t)b * kN * 3;
    const float* pb = pf  + (size_t)b * kN * 3;
    split_hilo_kernel<<<g1d(kN * 32, 256), 256, 0, stream>>>(xb, hib, lob, kN, 3, 32);
    sqnorm_kernel<<<g1d(kN, 256), 256, 0, stream>>>(xb, sq, kN, 3);
    gram_dist_kernel<<<ggram, 256, 0, stream>>>(hib, lob, sq, distY, kN, 32);
    int* ib = idxb + (size_t)b * kN * kK;
    topk_kernel<<<g1d(kN, 256), 256, 0, stream>>>(distY, ib, kN);
    gather_l1_kernel<<<g1d(kN * kK, 256), 256, 0, stream>>>(xb, pb, ib, hbuf, kN);
    dim3 gm1(64 / 16, Mrows / (16 * kGW));
    gemm_bn_lrelu_kernel<<<gm1, 256, 0, stream>>>(hbuf, wp1, bn[0][0], bn[0][1], bn[0][2],
                                                  bn[0][3], distY, Mrows, 32, 64);
    max_over_k_kernel<<<g1d(kN * 64, 256), 256, 0, stream>>>(distY, x1 + (size_t)b * kN * 64, kN, 64);

    // ---- EdgeConv 2..4 ----------------------------------------------------
    run_edge(b, x1, 64, 64,   wp2, 128, 64,  1, x2);
    run_edge(b, x2, 64, 64,   wp3, 128, 128, 2, x3);
    run_edge(b, x3, 128, 128, wp4, 256, 256, 3, x4);
  }

  // ---- conv5 on concat(x1..x4), then global pooling + MLP head ------------
  pack_xcat_kernel<<<g1d(kB * kN * 512, 256), 256, 0, stream>>>(x1, x2, x3, x4, xcat, kB * kN);
  dim3 g5(512 / 16, (kB * kN) / (16 * kGW));
  gemm_bn_lrelu_kernel<<<g5, 256, 0, stream>>>(xcat, wp5, bn[4][0], bn[4][1], bn[4][2], bn[4][3],
                                               y5, kB * kN, 512, 512);
  global_pool_kernel<<<g1d(kB * 512, 256), 256, 0, stream>>>(y5, zb, kN);
  fc_kernel<<<g1d(kB * 512, 256), 256, 0, stream>>>(zb, l1w, nullptr, bn[5][0], bn[5][1],
                                                    bn[5][2], bn[5][3], z1, kB, 1024, 512, 1);
  fc_kernel<<<g1d(kB * 256, 256), 256, 0, stream>>>(z1, l2w, l2b, bn[6][0], bn[6][1],
                                                    bn[6][2], bn[6][3], z2, kB, 512, 256, 1);
  fc_kernel<<<g1d(kB * 16, 64), 64, 0, stream>>>(z2, l3w, l3b, nullptr, nullptr, nullptr,
                                                 nullptr, out, kB, 256, 16, 0);
}
